// HIB_model_66769561583681
// MI455X (gfx1250) — matrix operand order repaired and checked by reading, not compiled, not run on hardware
//
#include <hip/hip_runtime.h>
#include <hip/hip_bf16.h>

typedef __attribute__((ext_vector_type(16))) _Float16 v16h;
typedef __attribute__((ext_vector_type(8)))  float    v8f;

#define HIB_B    2048
#define HIB_PTS  2048       // N*2 fused point stream per batch element
#define BETA_    0.001f
#define VAR_MIN_ 1e-7f

__device__ __forceinline__ float sp_(float x) {
  // softplus, overflow-safe (accurate version, used in cold epilogue)
  return (x > 20.0f) ? x : log1pf(__expf(x));
}

__device__ __forceinline__ float sp_fast_(float x) {
  // branchless fast softplus for the hot loop: 2 TRANS + add + cndmask
  float t = __logf(1.0f + __expf(x));
  return (x > 20.0f) ? x : t;
}

__global__ __launch_bounds__(256) void hib_phase1(
    const float* __restrict__ batch, const int* __restrict__ labels,
    const float* __restrict__ eps1,  const float* __restrict__ eps2,
    const float* __restrict__ W1,  const float* __restrict__ b1,
    const float* __restrict__ W2,  const float* __restrict__ b2,
    const float* __restrict__ Wmu, const float* __restrict__ bmu,
    const float* __restrict__ Wv,  const float* __restrict__ bv,
    const float* __restrict__ sw,  const float* __restrict__ sb,
    float* __restrict__ out_partial)
{
  __shared__ float sPart[8][4][8];   // [wave][group][z]
  __shared__ float sMu[2][8];
  __shared__ float sVar[2][8];
  __shared__ float sZ1[8][8];
  __shared__ float sZ2[8][8];
  __shared__ float sKL[16];
  __shared__ float sLoss[64];

  const int b    = blockIdx.x;
  const int tid  = threadIdx.x;
  const int lane = tid & 31;
  const int wave = tid >> 5;
  const int lh   = (lane >> 4) & 1;   // lane half (0: lanes 0-15, 1: 16-31)
  const int lr   = lane & 15;
  const bool isVarLane = (lh != 0);

  // ---- weights in WMMA A layout, K-permuted by involution P ----
  // P swaps true-feature blocks [8..15] <-> [16..23]; slot e -> true k =
  // e + 8*lh + (e>=8 ? 8 : 0), matching the B operands.
  v16h a1_0, a1_1, a2_0, a2_1, af;
  #pragma unroll
  for (int e = 0; e < 16; ++e) {
    const int k = e + 8 * lh + ((e >= 8) ? 8 : 0);
    a1_0[e] = (_Float16)W1[k * 32 + lr];
    a1_1[e] = (_Float16)W1[k * 32 + lr + 16];
    a2_0[e] = (_Float16)W2[k * 32 + lr];
    a2_1[e] = (_Float16)W2[k * 32 + lr + 16];
    af[e]   = (_Float16)((lr < 8) ? Wmu[k * 8 + lr] : Wv[k * 8 + (lr - 8)]);
  }
  // biases pre-broadcast into WMMA C layout: C vgpr i <-> out-feature i+8*lh
  v8f c1_0, c1_1, c2_0, c2_1, cf;
  #pragma unroll
  for (int i = 0; i < 8; ++i) {
    c1_0[i] = b1[i + 8 * lh];
    c1_1[i] = b1[16 + i + 8 * lh];
    c2_0[i] = b2[i + 8 * lh];
    c2_1[i] = b2[16 + i + 8 * lh];
    cf[i]   = lh ? bv[i] : bmu[i];
  }

  float acc[8];                       // lanes<16: sum mu ; lanes>=16: sum 1/var
  #pragma unroll
  for (int i = 0; i < 8; ++i) acc[i] = 0.0f;

  const float* xb = batch + (size_t)b * (32u * HIB_PTS);

  // 128 tiles of 16 points each; 8 waves -> 16 tiles/wave, loop fully uniform
  for (int t = wave; t < HIB_PTS / 16; t += 8) {
    const int p = t * 16 + lr;        // fused point index (n*2 + pair)
    // B-matrix x^T tile under permutation P: slot e -> true feature
    //   k = e + 8*lh + (e>=8 ? 8 : 0)   ({0..7,16..23} / {8..15,24..31})
    v16h bx;
    #pragma unroll
    for (int e = 0; e < 16; ++e) {
      const int k = e + 8 * lh + ((e >= 8) ? 8 : 0);
      bx[e] = (_Float16)xb[(size_t)k * HIB_PTS + p];
    }

    // layer 1 (bias folded into C)
    v8f d0 = __builtin_amdgcn_wmma_f32_16x16x32_f16(false, a1_0, false, bx, (short)0, c1_0, false, false);
    v8f d1 = __builtin_amdgcn_wmma_f32_16x16x32_f16(false, a1_1, false, bx, (short)0, c1_1, false, false);

    // D -> next-layer B is lane-local under permutation P:
    // slot e<8 <- D0[e], slot e>=8 <- D1[e-8]  (identical for both halves).
    // Convert first (cvt output is canonical), then relu in f16 so the
    // maxes need no canonicalize and pack into v_pk_max_num_f16.
    v16h h1;
    #pragma unroll
    for (int i = 0; i < 8; ++i) {
      h1[i]     = (_Float16)d0[i];
      h1[i + 8] = (_Float16)d1[i];
    }
    #pragma unroll
    for (int i = 0; i < 16; ++i)
      h1[i] = __builtin_fmaxf16(h1[i], (_Float16)0.0f);

    // layer 2
    v8f e0 = __builtin_amdgcn_wmma_f32_16x16x32_f16(false, a2_0, false, h1, (short)0, c2_0, false, false);
    v8f e1 = __builtin_amdgcn_wmma_f32_16x16x32_f16(false, a2_1, false, h1, (short)0, c2_1, false, false);

    v16h h2;
    #pragma unroll
    for (int i = 0; i < 8; ++i) {
      h2[i]     = (_Float16)e0[i];
      h2[i + 8] = (_Float16)e1[i];
    }
    #pragma unroll
    for (int i = 0; i < 16; ++i)
      h2[i] = __builtin_fmaxf16(h2[i], (_Float16)0.0f);

    // packed Wmu/Wv head: lanes<16 get mu[0..7], lanes>=16 get preVar[0..7]
    v8f df = __builtin_amdgcn_wmma_f32_16x16x32_f16(false, af, false, h2, (short)0, cf, false, false);

    // branchless tail: all lanes compute both candidates, one cndmask picks
    #pragma unroll
    for (int i = 0; i < 8; ++i) {
      float v   = df[i];
      float var = fmaxf(sp_fast_(v), VAR_MIN_);
      float iv  = __builtin_amdgcn_rcpf(var);
      acc[i] += isVarLane ? iv : v;
    }
  }

  // reduce lanes with same (pair parity, half): xor 2,4,8
  #pragma unroll
  for (int i = 0; i < 8; ++i) {
    acc[i] += __shfl_xor(acc[i], 2, 32);
    acc[i] += __shfl_xor(acc[i], 4, 32);
    acc[i] += __shfl_xor(acc[i], 8, 32);
  }
  if ((lane & 14) == 0) {                 // lanes 0,1,16,17 hold group sums
    const int g = (lane & 1) | (lh << 1); // 0:mu0 1:mu1 2:iv0 3:iv1
    #pragma unroll
    for (int i = 0; i < 8; ++i) sPart[wave][g][i] = acc[i];
  }
  __syncthreads();

  if (tid < 32) {
    const int g = tid >> 3, z = tid & 7;
    float s = 0.0f;
    #pragma unroll
    for (int w = 0; w < 8; ++w) s += sPart[w][g][z];
    if (g < 2) sMu[g][z] = s;             // mu_pog = sum(mu)
    else       sVar[g - 2][z] = 1.0f / s; // var_pog = 1/sum(1/var)
  }
  __syncthreads();

  if (tid < 64) {                         // rsample z1, z2   [S=8, Z=8]
    const int si = tid >> 3, z = tid & 7;
    sZ1[si][z] = sMu[0][z] + sqrtf(sVar[0][z]) * eps1[(size_t)b * 64 + si * 8 + z];
    sZ2[si][z] = sMu[1][z] + sqrtf(sVar[1][z]) * eps2[(size_t)b * 64 + si * 8 + z];
  } else if (tid < 80) {                  // KL elements, both streams
    const int q = tid - 64, pr = q >> 3, z = q & 7;
    float m = sMu[pr][z], v = sVar[pr][z];
    sKL[q] = -0.5f * logf(v) + 0.5f * (v + m * m) - 0.5f;
  }
  __syncthreads();

  if (tid < 64) {                         // 8x8 pairwise distances
    const int s1 = tid >> 3, s2 = tid & 7;
    float d2 = 0.0f;
    #pragma unroll
    for (int z = 0; z < 8; ++z) {
      float d = sZ1[s1][z] - sZ2[s2][z];
      d2 += d * d;
    }
    float dist = sqrtf(d2);
    float a = sp_(sw[0]);
    // -log(sigmoid(-a*d + sb)) == softplus(a*d - sb)
    sLoss[tid] = sp_(a * dist - sb[0]);
  }
  __syncthreads();

  if (tid == 0) {
    float kls = 0.0f;
    for (int i = 0; i < 16; ++i) kls += sKL[i];
    float ls = 0.0f;
    for (int i = 0; i < 64; ++i) ls += sLoss[i];
    float lab = (float)labels[b];
    out_partial[b] = BETA_ * kls / (float)(HIB_B * 8)
                   + lab * ls / (64.0f * (float)HIB_B);
  }
}

__global__ __launch_bounds__(256) void hib_final(const float* __restrict__ part,
                                                 float* __restrict__ out)
{
  __shared__ float red[256];
  float s = 0.0f;
  for (int i = threadIdx.x; i < HIB_B; i += 256) s += part[i];
  red[threadIdx.x] = s;
  __syncthreads();
  for (int off = 128; off > 0; off >>= 1) {
    if ((int)threadIdx.x < off) red[threadIdx.x] += red[threadIdx.x + off];
    __syncthreads();
  }
  if (threadIdx.x == 0) out[0] = red[0];
}

extern "C" void kernel_launch(void* const* d_in, const int* in_sizes, int n_in,
                              void* d_out, int out_size, void* d_ws, size_t ws_size,
                              hipStream_t stream) {
  const float* batch  = (const float*)d_in[0];
  const int*   labels = (const int*)  d_in[1];
  const float* eps1   = (const float*)d_in[2];
  const float* eps2   = (const float*)d_in[3];
  const float* W1  = (const float*)d_in[4];
  const float* b1  = (const float*)d_in[5];
  const float* W2  = (const float*)d_in[6];
  const float* b2  = (const float*)d_in[7];
  const float* Wmu = (const float*)d_in[8];
  const float* bmu = (const float*)d_in[9];
  const float* Wv  = (const float*)d_in[10];
  const float* bv  = (const float*)d_in[11];
  const float* sw  = (const float*)d_in[12];
  const float* sb  = (const float*)d_in[13];

  float* part = (float*)d_ws;   // HIB_B floats of per-batch partial loss

  hib_phase1<<<HIB_B, 256, 0, stream>>>(batch, labels, eps1, eps2,
                                        W1, b1, W2, b2, Wmu, bmu, Wv, bv,
                                        sw, sb, part);
  hib_final<<<1, 256, 0, stream>>>(part, (float*)d_out);
}
